// MultiHeadSparseAttention_87316685127936
// MI455X (gfx1250) — matrix operand order repaired and checked by reading, not compiled, not run on hardware
//
#include <hip/hip_runtime.h>
#include <hip/hip_bf16.h>

// ---------------------------------------------------------------------------
// Multi-head sparsemax attention for MI455X (gfx1250, wave32, WMMA).
//   0) cvt_f32_bf16        : one-time bf16 conversion of activations + weights
//   1) gemm_bf16<1>/<2> x3 : Q/K projections -> [B,H,S,HD]; V -> [B,H,HD,S]
//                            (A-panel staged in LDS, 4 N-tiles per wave)
//   2) attention_sparsemax : fused QK^T -> sparsemax (bisection) -> P.V
//                            scores live entirely in LDS (128KB dynamic)
//   3) gemm_bf16<0>        : output projection -> f32 [4096,512]
// All matmuls: v_wmma_f32_16x16x32_bf16 (f32 accumulate).
// ---------------------------------------------------------------------------

#define DEV __device__ __forceinline__

typedef __attribute__((ext_vector_type(16))) __bf16 v16bf;
typedef __attribute__((ext_vector_type(8)))  float  v8f;

union BF16Frag { v16bf v; unsigned short u[16]; };

static constexpr int BATCH    = 2;
static constexpr int SEQ      = 2048;
static constexpr int DMODEL   = 512;
static constexpr int NHEADS   = 8;
static constexpr int HDIM     = 64;
static constexpr int MROWS    = BATCH * SEQ;          // 4096
static constexpr float SCALE  = 0.125f;               // 64^-0.5
static constexpr int LDSTRIDE = 2052;                 // 2048 + pad (stride%64==4)

DEV unsigned short f2bf(float f) {                    // RNE float -> bf16 bits
    unsigned int u = __float_as_uint(f);
    u += 0x7FFFu + ((u >> 16) & 1u);
    return (unsigned short)(u >> 16);
}

DEV v8f wmma_bf16(const BF16Frag& a, const BF16Frag& b, v8f c) {
    return __builtin_amdgcn_wmma_f32_16x16x32_bf16(
        /*neg_a=*/false, a.v, /*neg_b=*/false, b.v,
        /*c_mod=*/(short)0, c, /*reuse_a=*/false, /*reuse_b=*/false);
}

// ---------------------------------------------------------------------------
// Elementwise f32 -> bf16 (vectorized x4).  n must be a multiple of 4.
// ---------------------------------------------------------------------------
__global__ __launch_bounds__(256)
void cvt_f32_bf16(const float* __restrict__ s, unsigned short* __restrict__ d, int n4) {
    const int i = blockIdx.x * 256 + threadIdx.x;
    if (i < n4) {
        const float4 f = ((const float4*)s)[i];
        uint2 o;
        o.x = (unsigned)f2bf(f.x) | ((unsigned)f2bf(f.y) << 16);
        o.y = (unsigned)f2bf(f.z) | ((unsigned)f2bf(f.w) << 16);
        ((uint2*)d)[i] = o;
    }
}

// ---------------------------------------------------------------------------
// C[4096 x 512] = A[4096 x 512] @ W[512 x 512]^T + bias   (A, W in bf16 bits)
// Block = 256 threads (8 waves). A-panel (16 x 512 bf16, 16KB) staged in LDS.
// Wave w computes a 16 x 64 strip: N-tiles w*64 .. w*64+63 (4 accumulators).
// OUTMODE: 0 = f32 [M,N], 1 = bf16 [B,H,S,HD], 2 = bf16 [B,H,HD,S] (V^T).
// ---------------------------------------------------------------------------
template <int OUTMODE>
__global__ __launch_bounds__(256)
void gemm_bf16(const unsigned short* __restrict__ A,
               const unsigned short* __restrict__ W,
               const float* __restrict__ bias, void* __restrict__ C_) {
    constexpr int K = DMODEL;
    __shared__ unsigned short sA[16 * K];               // 16 KB

    const int tid  = threadIdx.x;
    const int wave = tid >> 5;
    const int lane = tid & 31;
    const int hl   = lane >> 4;
    const int l16  = lane & 15;
    const int m0   = blockIdx.x * 16;

    // Cooperative A-panel load: 1024 x 16B chunks, coalesced.
    {
        const uint4* src = (const uint4*)(A + (size_t)m0 * K);
        uint4*       dst = (uint4*)sA;
        for (int c = tid; c < 16 * K / 8; c += 256) dst[c] = src[c];
    }
    __syncthreads();

    v8f acc[4] = {{}, {}, {}, {}};
    const int n0 = wave * 64;

    for (int k0 = 0; k0 < K; k0 += 32) {
        BF16Frag a;
        // A fragment from LDS: element e -> K = k0 + (e<8 ? e : e+8) + hl*8
        const unsigned short* arow = sA + l16 * K + k0 + hl * 8;
#pragma unroll
        for (int e = 0; e < 8; ++e) { a.u[e] = arow[e]; a.u[8 + e] = arow[16 + e]; }
#pragma unroll
        for (int nt = 0; nt < 4; ++nt) {
            BF16Frag b;                                  // B[k][n] = W[n][k]
            const unsigned short* wrow =
                W + (size_t)(n0 + nt * 16 + l16) * K + k0 + hl * 8;
#pragma unroll
            for (int e = 0; e < 16; ++e) b.u[e] = wrow[(e < 8) ? e : e + 8];
            acc[nt] = wmma_bf16(a, b, acc[nt]);
        }
    }
    // C layout: VGPR r -> row m0 + r + hl*8, col ntile + l16
#pragma unroll
    for (int nt = 0; nt < 4; ++nt) {
#pragma unroll
        for (int r = 0; r < 8; ++r) {
            const int row = m0 + r + hl * 8;
            const int col = n0 + nt * 16 + l16;
            const float val = acc[nt][r] + bias[col];
            if constexpr (OUTMODE == 0) {
                ((float*)C_)[(size_t)row * DMODEL + col] = val;
            } else {
                const int bb = row / SEQ, s = row % SEQ;
                const int h  = col / HDIM, d = col % HDIM;
                const size_t idx = (OUTMODE == 1)
                    ? (((size_t)(bb * NHEADS + h) * SEQ) + s) * HDIM + d
                    : (((size_t)(bb * NHEADS + h) * HDIM) + d) * SEQ + s;
                ((unsigned short*)C_)[idx] = f2bf(val);
            }
        }
    }
}

// ---------------------------------------------------------------------------
// Fused attention: one block = one (b,h) and one 16-query tile.
// 256 threads = 8 waves. Dynamic LDS: scores[16][LDSTRIDE] f32 + cred[4][16][16].
// Qh, Kh: [B,H,S,HD] bf16;  Vt: [B,H,HD,S] bf16;  O: [B,S,H,HD] bf16.
// ---------------------------------------------------------------------------
__global__ __launch_bounds__(256)
void attention_sparsemax(const unsigned short* __restrict__ Qh,
                         const unsigned short* __restrict__ Kh,
                         const unsigned short* __restrict__ Vt,
                         unsigned short* __restrict__ O) {
    extern __shared__ char smem[];
    float* sc   = (float*)smem;                 // [16][LDSTRIDE]
    float* cred = sc + 16 * LDSTRIDE;           // [4][16][16]

    const int wave = threadIdx.x >> 5;
    const int lane = threadIdx.x & 31;
    const int hl   = lane >> 4;
    const int l16  = lane & 15;
    const int bh = blockIdx.x;                  // b*NHEADS + h
    const int b  = bh / NHEADS, h = bh % NHEADS;
    const int q0 = blockIdx.y * 16;

    const unsigned short* Qb = Qh + (size_t)bh * SEQ * HDIM;
    const unsigned short* Kb = Kh + (size_t)bh * SEQ * HDIM;
    const unsigned short* Vb = Vt + (size_t)bh * HDIM * SEQ;

    // ---- Phase A: scores = (Q @ K^T) * SCALE, streamed into LDS -----------
    BF16Frag aQ0, aQ1;                          // Q rows, HD 0..31 and 32..63
    {
        const unsigned short* qrow = Qb + (size_t)(q0 + l16) * HDIM;
#pragma unroll
        for (int e = 0; e < 16; ++e) {
            const int kk = ((e < 8) ? e : e + 8) + hl * 8;
            aQ0.u[e] = qrow[kk];
            aQ1.u[e] = qrow[32 + kk];
        }
    }
    for (int kt = wave; kt < SEQ / 16; kt += 8) {
        const int key0 = kt * 16;
        BF16Frag b0, b1;                        // K^T: B[d][n] = K[key0+n][d]
        const unsigned short* krow = Kb + (size_t)(key0 + l16) * HDIM + hl * 8;
#pragma unroll
        for (int e = 0; e < 16; ++e) { const int kk = (e < 8) ? e : e + 8; b0.u[e] = krow[kk]; b1.u[e] = krow[32 + kk]; }
        v8f acc = {};
        acc = wmma_bf16(aQ0, b0, acc);
        acc = wmma_bf16(aQ1, b1, acc);
#pragma unroll
        for (int r = 0; r < 8; ++r)
            sc[(size_t)(r + hl * 8) * LDSTRIDE + key0 + l16] = acc[r] * SCALE;
    }
    __syncthreads();

    // ---- Phase B: sparsemax per row via bisection on tau ------------------
    // f(tau) = sum(max(x - tau, 0)) is monotone, root in [max-1, max].
#pragma unroll
    for (int rr = 0; rr < 2; ++rr) {
        const int row = wave + rr * 8;          // rows disjoint across waves
        float* srow = sc + (size_t)row * LDSTRIDE;
        float v[64];
#pragma unroll
        for (int j = 0; j < 64; ++j) v[j] = srow[lane + 32 * j];

        float m = -3.0e38f;
#pragma unroll
        for (int j = 0; j < 64; ++j) m = fmaxf(m, v[j]);
#pragma unroll
        for (int o = 16; o > 0; o >>= 1) m = fmaxf(m, __shfl_xor(m, o, 32));

        float lo = m - 1.0f, hi = m;
        for (int it = 0; it < 32; ++it) {
            const float mid = 0.5f * (lo + hi);
            float s = 0.0f;
#pragma unroll
            for (int j = 0; j < 64; ++j) s += fmaxf(v[j] - mid, 0.0f);
#pragma unroll
            for (int o = 16; o > 0; o >>= 1) s += __shfl_xor(s, o, 32);
            if (s >= 1.0f) lo = mid; else hi = mid;   // uniform across wave
        }
        const float tau = 0.5f * (lo + hi);
#pragma unroll
        for (int j = 0; j < 64; ++j) srow[lane + 32 * j] = fmaxf(v[j] - tau, 0.0f);
    }
    __syncthreads();

    // ---- Phase C: out = P @ V (16 x 64). Wave w: HD tile (w&3), key half (w>>2).
    const int t     = wave & 3;
    const int khalf = wave >> 2;
    v8f acc = {};
    for (int c = 0; c < 32; ++c) {
        const int kc = (khalf * 32 + c) * 32;   // key-chunk base
        BF16Frag a, bv;
#pragma unroll
        for (int e = 0; e < 16; ++e) {          // A = P tile from LDS
            const int key = kc + ((e < 8) ? e : e + 8) + hl * 8;
            a.u[e] = f2bf(sc[(size_t)l16 * LDSTRIDE + key]);
        }
        // B[k][n] = V[key][hd]; Vt layout makes this contiguous per lane.
        const unsigned short* vrow =
            Vb + (size_t)(t * 16 + l16) * SEQ + kc + hl * 8;
#pragma unroll
        for (int e = 0; e < 16; ++e) bv.u[e] = vrow[(e < 8) ? e : e + 8];
        acc = wmma_bf16(a, bv, acc);
    }
    float* ct = cred + t * 256;
    if (khalf == 1) {
#pragma unroll
        for (int r = 0; r < 8; ++r) ct[(r + hl * 8) * 16 + l16] = acc[r];
    }
    __syncthreads();
    if (khalf == 0) {
#pragma unroll
        for (int r = 0; r < 8; ++r) {
            const float val = acc[r] + ct[(r + hl * 8) * 16 + l16];
            const int row = q0 + r + hl * 8;        // query index s
            const int hd  = t * 16 + l16;
            // Output staged as [B, S, H, HD] == plain [4096, 512] bf16 matrix
            O[((size_t)(b * SEQ + row) * DMODEL) + h * HDIM + hd] = f2bf(val);
        }
    }
}

// ---------------------------------------------------------------------------
extern "C" void kernel_launch(void* const* d_in, const int* in_sizes, int n_in,
                              void* d_out, int out_size, void* d_ws, size_t ws_size,
                              hipStream_t stream) {
    const float* q  = (const float*)d_in[0];
    const float* k  = (const float*)d_in[1];
    const float* Wq = (const float*)d_in[2];
    const float* bq = (const float*)d_in[3];
    const float* Wk = (const float*)d_in[4];
    const float* bk = (const float*)d_in[5];
    const float* Wv = (const float*)d_in[6];
    const float* bv = (const float*)d_in[7];
    const float* Wo = (const float*)d_in[8];
    const float* bo = (const float*)d_in[9];

    const size_t NEL = (size_t)MROWS * DMODEL;      // 2,097,152 elements
    const size_t WEL = (size_t)DMODEL * DMODEL;     //   262,144 elements
    unsigned short* p    = (unsigned short*)d_ws;   // carve ~27 MB of d_ws
    unsigned short* qbf  = p;  p += NEL;
    unsigned short* kbf  = p;  p += NEL;
    unsigned short* Wqb  = p;  p += WEL;
    unsigned short* Wkb  = p;  p += WEL;
    unsigned short* Wvb  = p;  p += WEL;
    unsigned short* Wob  = p;  p += WEL;
    unsigned short* Qh   = p;  p += NEL;
    unsigned short* Kh   = p;  p += NEL;
    unsigned short* Vt   = p;  p += NEL;
    unsigned short* Oc   = p;  p += NEL;

    dim3 blk(256);
    // 0) one-time bf16 conversions
    cvt_f32_bf16<<<dim3((NEL / 4 + 255) / 256), blk, 0, stream>>>(q,  qbf, (int)(NEL / 4));
    cvt_f32_bf16<<<dim3((NEL / 4 + 255) / 256), blk, 0, stream>>>(k,  kbf, (int)(NEL / 4));
    cvt_f32_bf16<<<dim3((WEL / 4 + 255) / 256), blk, 0, stream>>>(Wq, Wqb, (int)(WEL / 4));
    cvt_f32_bf16<<<dim3((WEL / 4 + 255) / 256), blk, 0, stream>>>(Wk, Wkb, (int)(WEL / 4));
    cvt_f32_bf16<<<dim3((WEL / 4 + 255) / 256), blk, 0, stream>>>(Wv, Wvb, (int)(WEL / 4));
    cvt_f32_bf16<<<dim3((WEL / 4 + 255) / 256), blk, 0, stream>>>(Wo, Wob, (int)(WEL / 4));

    // 1) projections
    dim3 gproj(MROWS / 16, 1);
    gemm_bf16<1><<<gproj, blk, 0, stream>>>(qbf, Wqb, bq, Qh);
    gemm_bf16<1><<<gproj, blk, 0, stream>>>(kbf, Wkb, bk, Kh);
    gemm_bf16<2><<<gproj, blk, 0, stream>>>(kbf, Wvb, bv, Vt);

    // 2) fused sparsemax attention
    const size_t shm = (16 * (size_t)LDSTRIDE + 4 * 256) * sizeof(float); // ~135 KB
    (void)hipFuncSetAttribute((const void*)attention_sparsemax,
                              hipFuncAttributeMaxDynamicSharedMemorySize, (int)shm);
    attention_sparsemax<<<dim3(BATCH * NHEADS, SEQ / 16), blk, shm, stream>>>(Qh, Kh, Vt, Oc);

    // 3) output projection
    gemm_bf16<0><<<gproj, blk, 0, stream>>>(Oc, Wob, bo, (float*)d_out);
}
